// RWKV_TimeMix_76124000354320
// MI455X (gfx1250) — matrix operand order repaired and checked by reading, not compiled, not run on hardware
//
#include <hip/hip_runtime.h>
#include <hip/hip_bf16.h>

// ---------------------------------------------------------------------------
// RWKV-style time-mix for MI455X (gfx1250, wave32).
//   r,k,v GEMMs + output GEMM   : bf16 WMMA 16x16x32 (ternary weights exact in
//                                 bf16; per-row scale applied in f32 epilogue)
//   tiles staged into LDS       : Tensor Data Mover (tensor_load_to_lds) with
//                                 LDS padding (80B row stride, conflict-free)
//   decay recurrence            : 3-pass chunked scan (64 chunks x 64 steps)
// Workspace use: ~278 MB (x_bf16, wq, scales, r/k/v buffers, carries, normed).
// ---------------------------------------------------------------------------

typedef unsigned int u32;
typedef u32   u32x4  __attribute__((ext_vector_type(4)));
typedef int   i32x4  __attribute__((ext_vector_type(4)));
typedef int   i32x8  __attribute__((ext_vector_type(8)));
typedef __bf16 v16bf __attribute__((ext_vector_type(16)));
typedef float  v8f   __attribute__((ext_vector_type(8)));

#define D_MODEL 1024
#define T_LEN   4096
#define B_SZ    4
#define M_TOK   (B_SZ * T_LEN)          // 16384 tokens
#define NCHUNK  64
#define CHUNK   64                      // NCHUNK*CHUNK == T_LEN

// ---------------------------------------------------------------------------
// TDM: issue a 2D tile load (tile_k x tile_rows bf16 elements) into LDS.
// D# group0: count=1, lds_addr, global_addr(57b), type=2.
// D# group1: data_size=1(2B), pad_enable, pad_interval=3 (every 64B),
//            pad_amount=3 (16B)  ->  LDS row stride = 80 bytes.
// This toolchain exposes the 6-arg builtin:
//   (u32x4 g0, i32x8 g1, i32x4 g2, i32x4 g3, i32x8 g4, i32 cpol)
// ---------------------------------------------------------------------------
__device__ __forceinline__ void tdm_load_tile_2d(u32 lds_byte, const void* gptr,
                                                 u32 tile_k, u32 tile_rows,
                                                 u32 tensor_k, u32 tensor_rows,
                                                 u32 row_stride_elems) {
  unsigned long long ga = (unsigned long long)(uintptr_t)gptr;
  u32x4 g0;
  g0.x = 1u;                                    // count = 1 (valid descriptor)
  g0.y = lds_byte;                              // LDS byte address
  g0.z = (u32)ga;                               // global_addr[31:0]
  g0.w = (u32)((ga >> 32) & 0x01FFFFFFull)      // global_addr[56:32]
       | (2u << 30);                            // type = 2 ("image")
  i32x8 g1;
  g1[0] = (int)((1u << 16)                      // data_size = 1 -> 2 bytes
              | (1u << 20)                      // pad_enable
              | (3u << 22)                      // pad_interval = 16 DWORDs (64B)
              | (3u << 25));                    // pad_amount   = 4 DWORDs (16B)
  g1[1] = (int)((tensor_k & 0xFFFFu) << 16);                    // tensor_dim0 lo
  g1[2] = (int)((tensor_k >> 16) | ((tensor_rows & 0xFFFFu) << 16));
  g1[3] = (int)((tensor_rows >> 16) | (tile_k << 16));          // tile_dim0
  g1[4] = (int)(tile_rows & 0xFFFFu);                           // tile_dim1
  g1[5] = (int)row_stride_elems;                                // dim0 stride
  g1[6] = 0;
  g1[7] = 0;
  i32x4 z4 = {0, 0, 0, 0};
  i32x8 z8 = {0, 0, 0, 0, 0, 0, 0, 0};
  __builtin_amdgcn_tensor_load_to_lds(g0, g1, z4, z4, z8, 0);
}

// Fragment loads from the padded LDS tiles (row stride 80B, 16B aligned).
// A (16x32 bf16): lane group g holds K = {g*8 .. g*8+7} and {16+g*8 .. 16+g*8+7}.
__device__ __forceinline__ v16bf ld_frag_a(const unsigned char* s, u32 row_byte, u32 g) {
  union { uint4 u[2]; v16bf v; } f;
  f.u[0] = *(const uint4*)(s + row_byte + g * 16);
  f.u[1] = *(const uint4*)(s + row_byte + 32 + g * 16);
  return f.v;
}
// B (32x16 bf16): lane group g holds column idx, K = {g*8 .. g*8+7} and {16+g*8 .. 16+g*8+7}.
__device__ __forceinline__ v16bf ld_frag_b(const unsigned char* s, u32 row_byte, u32 g) {
  union { uint4 u[2]; v16bf v; } f;
  f.u[0] = *(const uint4*)(s + row_byte + g * 16);
  f.u[1] = *(const uint4*)(s + row_byte + 32 + g * 16);
  return f.v;
}

// ---------------------------------------------------------------------------
// GEMM: out[m,n] = scale[n] * sum_k A[m,k] * Wq[n,k]     (mode 1: sigmoid)
// Block = 256 threads (8 waves), block tile 128x64, wave tile 32x32.
// LDS: A tile 128 rows * 80B = 10240B, B tile 64 rows * 80B = 5120B.
// ---------------------------------------------------------------------------
__global__ __launch_bounds__(256)
void gemm_bf16_wmma(const __bf16* __restrict__ A, const __bf16* __restrict__ Wq,
                    const float* __restrict__ scale, float* __restrict__ out,
                    int M, int mode) {
  const int tid  = threadIdx.x;
  const int wid  = tid >> 5;
  const int lane = tid & 31;
  const int idx  = lane & 15;
  const u32 g    = (u32)(lane >> 4);
  const int waveM = wid & 3;          // 4 waves along M
  const int waveN = wid >> 2;         // 2 waves along N
  const int Mbase = blockIdx.y * 128;
  const int Nbase = blockIdx.x * 64;

  __shared__ alignas(16) unsigned char smem[15360];
  const u32 lds0 = (u32)(uintptr_t)&smem[0];

  v8f acc[2][2] = {};

  const u32 aoff0 = (u32)((waveM * 32 + 0 * 16 + idx) * 80);
  const u32 aoff1 = (u32)((waveM * 32 + 1 * 16 + idx) * 80);
  const u32 boff0 = 10240u + (u32)((waveN * 32 + 0 * 16 + idx) * 80);
  const u32 boff1 = 10240u + (u32)((waveN * 32 + 1 * 16 + idx) * 80);

  for (int kb = 0; kb < D_MODEL / 32; ++kb) {
    if (wid == 0) {
      // X tile: 128 rows x 32 K-elements from A[Mbase.., kb*32..]
      tdm_load_tile_2d(lds0, A + (size_t)Mbase * D_MODEL + kb * 32,
                       32u, 128u, (u32)D_MODEL, (u32)M, (u32)D_MODEL);
      // W tile: 64 rows (output cols) x 32 K-elements
      tdm_load_tile_2d(lds0 + 10240u, Wq + (size_t)Nbase * D_MODEL + kb * 32,
                       32u, 64u, (u32)D_MODEL, (u32)D_MODEL, (u32)D_MODEL);
      __builtin_amdgcn_s_wait_tensorcnt(0);
    }
    __syncthreads();

    v16bf a0 = ld_frag_a(smem, aoff0, g);
    v16bf a1 = ld_frag_a(smem, aoff1, g);
    v16bf b0 = ld_frag_b(smem, boff0, g);
    v16bf b1 = ld_frag_b(smem, boff1, g);

    acc[0][0] = __builtin_amdgcn_wmma_f32_16x16x32_bf16(false, a0, false, b0,
                                                        (short)0, acc[0][0], false, false);
    acc[0][1] = __builtin_amdgcn_wmma_f32_16x16x32_bf16(false, a0, false, b1,
                                                        (short)0, acc[0][1], false, false);
    acc[1][0] = __builtin_amdgcn_wmma_f32_16x16x32_bf16(false, a1, false, b0,
                                                        (short)0, acc[1][0], false, false);
    acc[1][1] = __builtin_amdgcn_wmma_f32_16x16x32_bf16(false, a1, false, b1,
                                                        (short)0, acc[1][1], false, false);
    __syncthreads();
  }

  // Epilogue: C/D layout — VGPR r: lanes 0-15 -> row r, lanes 16-31 -> row r+8.
  for (int mi = 0; mi < 2; ++mi) {
    const int rbase = Mbase + waveM * 32 + mi * 16 + (int)g * 8;
    for (int ni = 0; ni < 2; ++ni) {
      const int col = Nbase + waveN * 32 + ni * 16 + idx;
      const float sc = scale[col];
      for (int r = 0; r < 8; ++r) {
        float v = acc[mi][ni][r] * sc;
        if (mode == 1) v = 1.0f / (1.0f + __expf(-v));
        out[(size_t)(rbase + r) * D_MODEL + col] = v;
      }
    }
  }
}

// ---------------------------------------------------------------------------
// Elementwise / reduction kernels
// ---------------------------------------------------------------------------
__global__ void cvt_bf16_kernel(const float* __restrict__ x, __bf16* __restrict__ xb, int n) {
  int i = blockIdx.x * 256 + threadIdx.x;
  if (i < n) xb[i] = (__bf16)x[i];
}

// One block per weight row. scale = max(mean|w|, 1e-5); wq = clip(round(w/scale),-1,1).
__global__ __launch_bounds__(256)
void quantize_w_kernel(const float* __restrict__ Wr, const float* __restrict__ Wk,
                       const float* __restrict__ Wv, const float* __restrict__ Wo,
                       __bf16* __restrict__ wq, float* __restrict__ scales) {
  const int rowg = blockIdx.x;             // 0..4095
  const int widx = rowg >> 10;
  const int row  = rowg & 1023;
  const float* W = (widx == 0) ? Wr : (widx == 1) ? Wk : (widx == 2) ? Wv : Wo;
  const float* wrow = W + (size_t)row * D_MODEL;

  __shared__ float red[256];
  float s = 0.0f;
  for (int i = threadIdx.x; i < D_MODEL; i += 256) s += fabsf(wrow[i]);
  red[threadIdx.x] = s;
  __syncthreads();
  for (int off = 128; off > 0; off >>= 1) {
    if (threadIdx.x < off) red[threadIdx.x] += red[threadIdx.x + off];
    __syncthreads();
  }
  float sc = fmaxf(red[0] * (1.0f / D_MODEL), 1e-5f);
  if (threadIdx.x == 0) scales[rowg] = sc;
  const float inv = 1.0f / sc;
  __bf16* outrow = wq + (size_t)rowg * D_MODEL;
  for (int i = threadIdx.x; i < D_MODEL; i += 256) {
    float q = rintf(wrow[i] * inv);               // round-half-even, like jnp.round
    q = fminf(1.0f, fmaxf(-1.0f, q));
    outrow[i] = (__bf16)q;                        // exact in bf16
  }
}

__global__ void mul_kv_kernel(float* __restrict__ k, const float* __restrict__ v, int n) {
  int i = blockIdx.x * 256 + threadIdx.x;
  if (i < n) k[i] = k[i] * v[i];
}

__device__ __forceinline__ float sigmoidf(float x) { return 1.0f / (1.0f + __expf(-x)); }

// Pass A: per (b, chunk, d) local recurrence from zero; store chunk-end state.
__global__ void scan_passA(const float* __restrict__ kv, const float* __restrict__ decay,
                           float* __restrict__ carry) {
  int t = blockIdx.x * 256 + threadIdx.x;        // 262144 threads
  int d = t & (D_MODEL - 1);
  int rest = t >> 10;
  int c = rest & (NCHUNK - 1);
  int b = rest >> 6;
  const float dec = sigmoidf(decay[d]);
  const float* p = kv + ((size_t)(b * T_LEN + c * CHUNK)) * D_MODEL + d;
  float s = 0.0f;
  for (int j = 0; j < CHUNK; ++j) s = __fmaf_rn(dec, s, p[(size_t)j * D_MODEL]);
  carry[((size_t)(b * NCHUNK + c)) * D_MODEL + d] = s;
}

// Pass B: per (b, d) exclusive scan over chunks with ratio dec^CHUNK (in place).
__global__ void scan_passB(float* __restrict__ carry, const float* __restrict__ decay) {
  int t = blockIdx.x * 256 + threadIdx.x;        // 4096 threads
  int d = t & (D_MODEL - 1);
  int b = t >> 10;
  const float dec  = sigmoidf(decay[d]);
  const float decL = __powf(dec, (float)CHUNK);
  float run = 0.0f;
  for (int c = 0; c < NCHUNK; ++c) {
    size_t i = ((size_t)(b * NCHUNK + c)) * D_MODEL + d;
    float v = carry[i];
    carry[i] = run;                              // state entering chunk c
    run = __fmaf_rn(decL, run, v);
  }
}

// Pass C: recompute with carried-in state; y[t] = r[t] * state[t] (into r buffer).
__global__ void scan_passC(const float* __restrict__ kv, const float* __restrict__ carry,
                           const float* __restrict__ decay, float* __restrict__ r_y) {
  int t = blockIdx.x * 256 + threadIdx.x;
  int d = t & (D_MODEL - 1);
  int rest = t >> 10;
  int c = rest & (NCHUNK - 1);
  int b = rest >> 6;
  const float dec = sigmoidf(decay[d]);
  size_t base = ((size_t)(b * T_LEN + c * CHUNK)) * D_MODEL + d;
  float s = carry[((size_t)(b * NCHUNK + c)) * D_MODEL + d];
  for (int j = 0; j < CHUNK; ++j) {
    size_t i = base + (size_t)j * D_MODEL;
    s = __fmaf_rn(dec, s, kv[i]);
    r_y[i] = r_y[i] * s;
  }
}

// RMSNorm per token -> bf16 (WMMA A operand for the output GEMM).
__global__ __launch_bounds__(256)
void rmsnorm_bf16_kernel(const float* __restrict__ y, const float* __restrict__ lnw,
                         __bf16* __restrict__ out) {
  const int tok = blockIdx.x;
  const float* p = y + (size_t)tok * D_MODEL;
  __shared__ float red[256];
  float vals[4];
  float s = 0.0f;
  for (int j = 0; j < 4; ++j) {
    vals[j] = p[threadIdx.x + j * 256];
    s += vals[j] * vals[j];
  }
  red[threadIdx.x] = s;
  __syncthreads();
  for (int off = 128; off > 0; off >>= 1) {
    if (threadIdx.x < off) red[threadIdx.x] += red[threadIdx.x + off];
    __syncthreads();
  }
  const float inv = rsqrtf(red[0] * (1.0f / D_MODEL) + 1e-6f);
  for (int j = 0; j < 4; ++j) {
    int col = threadIdx.x + j * 256;
    out[(size_t)tok * D_MODEL + col] = (__bf16)(vals[j] * inv * lnw[col]);
  }
}

// ---------------------------------------------------------------------------
// Launch
// ---------------------------------------------------------------------------
extern "C" void kernel_launch(void* const* d_in, const int* in_sizes, int n_in,
                              void* d_out, int out_size, void* d_ws, size_t ws_size,
                              hipStream_t stream) {
  const float* x     = (const float*)d_in[0];
  const float* Wr    = (const float*)d_in[1];
  const float* Wk    = (const float*)d_in[2];
  const float* Wv    = (const float*)d_in[3];
  const float* Wo    = (const float*)d_in[4];
  const float* decay = (const float*)d_in[5];
  const float* lnw   = (const float*)d_in[6];
  float* out = (float*)d_out;

  char* ws = (char*)d_ws;
  __bf16* xb     = (__bf16*)(ws + 0);            //  32 MB  x in bf16
  __bf16* wq     = (__bf16*)(ws + 33554432);     //   8 MB  4 ternary weights (bf16)
  float*  scales = (float*) (ws + 41943040);     //  16 KB  per-row scales [4][1024]
  float*  rbuf   = (float*) (ws + 41959424);     //  64 MB  r, then y = r*state
  float*  kbuf   = (float*) (ws + 109068288);    //  64 MB  k, then kv = k*v
  float*  vbuf   = (float*) (ws + 176177152);    //  64 MB  v
  float*  carry  = (float*) (ws + 243286016);    //   1 MB  chunk carries
  __bf16* nb     = (__bf16*)(ws + 244334592);    //  32 MB  rmsnorm output (bf16)

  const int nElem = M_TOK * D_MODEL;             // 16,777,216

  cvt_bf16_kernel<<<nElem / 256, 256, 0, stream>>>(x, xb, nElem);
  quantize_w_kernel<<<4 * D_MODEL, 256, 0, stream>>>(Wr, Wk, Wv, Wo, wq, scales);

  dim3 ggrid(D_MODEL / 64, M_TOK / 128);         // (16, 128)
  gemm_bf16_wmma<<<ggrid, 256, 0, stream>>>(xb, wq + 0 * 1048576, scales + 0, rbuf, M_TOK, 1);
  gemm_bf16_wmma<<<ggrid, 256, 0, stream>>>(xb, wq + 1 * 1048576, scales + 1024, kbuf, M_TOK, 0);
  gemm_bf16_wmma<<<ggrid, 256, 0, stream>>>(xb, wq + 2 * 1048576, scales + 2048, vbuf, M_TOK, 0);

  mul_kv_kernel<<<nElem / 256, 256, 0, stream>>>(kbuf, vbuf, nElem);

  scan_passA<<<(B_SZ * NCHUNK * D_MODEL) / 256, 256, 0, stream>>>(kbuf, decay, carry);
  scan_passB<<<(B_SZ * D_MODEL) / 256, 256, 0, stream>>>(carry, decay);
  scan_passC<<<(B_SZ * NCHUNK * D_MODEL) / 256, 256, 0, stream>>>(kbuf, carry, decay, rbuf);

  rmsnorm_bf16_kernel<<<M_TOK, 256, 0, stream>>>(rbuf, lnw, nb);

  gemm_bf16_wmma<<<ggrid, 256, 0, stream>>>(nb, wq + 3 * 1048576, scales + 3072, out, M_TOK, 0);
}
